// TransLayer_32976758899135
// MI455X (gfx1250) — matrix operand order repaired and checked by reading, not compiled, not run on hardware
//
#include <hip/hip_runtime.h>

// ---------------------------------------------------------------------------
// Nystrom-attention TransLayer for gfx1250 (CDNA5, wave32, WMMA).
// ---------------------------------------------------------------------------

typedef __attribute__((ext_vector_type(16))) _Float16 v16h;
typedef __attribute__((ext_vector_type(8)))  float    v8f;
typedef __attribute__((ext_vector_type(2)))  float    v2f;

constexpr int BATCH = 2;
constexpr int SEQ   = 8192;
constexpr int DM    = 512;
constexpr int NH    = 8;
constexpr int DH    = 64;
constexpr int LM    = 256;          // landmarks
constexpr int BN    = BATCH * SEQ;  // 16384
constexpr int BH    = BATCH * NH;   // 16

#define DEVFN __device__ __forceinline__

// ------------------------------ WMMA wrappers ------------------------------
DEVFN v8f wmma16(v16h a, v16h b, v8f c) {
  return __builtin_amdgcn_wmma_f32_16x16x32_f16(false, a, false, b, (short)0, c,
                                                false, false);
}
DEVFN v8f wmma4(v2f a, v2f b, v8f c) {
  return __builtin_amdgcn_wmma_f32_16x16x4_f32(false, a, false, b, (short)0, c,
                                               false, false);
}

// A fragment, 16x32 f16, element(m,k) = p[m*ld + k]
DEVFN v16h load_a(const _Float16* p, int ld) {
  int lane = threadIdx.x & 31;
  int m = lane & 15, hi = lane >> 4;
  v16h a;
#pragma unroll
  for (int e = 0; e < 16; ++e) {
    int k = hi * 16 + e;
    a[e] = p[m * ld + k];
  }
  return a;
}
// B fragment for C = A @ Bt^T; Bt stored row-major [n][k]
DEVFN v16h load_bt(const _Float16* p, int ld) {
  int lane = threadIdx.x & 31;
  int n = lane & 15, hi = lane >> 4;
  v16h b;
#pragma unroll
  for (int e = 0; e < 16; ++e) b[e] = p[n * ld + hi * 16 + e];
  return b;
}
// B fragment for C = A @ B; B stored row-major [k][n]
DEVFN v16h load_b(const _Float16* p, int ld) {
  int lane = threadIdx.x & 31;
  int n = lane & 15, hi = lane >> 4;
  v16h b;
#pragma unroll
  for (int e = 0; e < 16; ++e) b[e] = p[(hi * 16 + e) * ld + n];
  return b;
}

// ------------------------------ small kernels ------------------------------
__global__ void f2h_kernel(const float* __restrict__ s, _Float16* __restrict__ d,
                           long n) {
  long i = (long)blockIdx.x * 256 + threadIdx.x;
  if (i < n) d[i] = (_Float16)s[i];
}

__global__ void layernorm_kernel(const float* __restrict__ x,
                                 const float* __restrict__ g,
                                 const float* __restrict__ be,
                                 _Float16* __restrict__ xnh) {
  __shared__ float red[256];
  long row = blockIdx.x;
  const float* p = x + row * DM;
  int t = threadIdx.x;
  float a = p[t], b2 = p[t + 256];
  red[t] = a + b2;
  __syncthreads();
  for (int s = 128; s > 0; s >>= 1) {
    if (t < s) red[t] += red[t + s];
    __syncthreads();
  }
  float mu = red[0] * (1.0f / DM);
  __syncthreads();
  float da = a - mu, db = b2 - mu;
  red[t] = da * da + db * db;
  __syncthreads();
  for (int s = 128; s > 0; s >>= 1) {
    if (t < s) red[t] += red[t + s];
    __syncthreads();
  }
  float rstd = rsqrtf(red[0] * (1.0f / DM) + 1e-5f);
  xnh[row * DM + t]       = (_Float16)(da * rstd * g[t] + be[t]);
  xnh[row * DM + t + 256] = (_Float16)(db * rstd * g[t + 256] + be[t + 256]);
}

__global__ void landmark_kernel(const _Float16* __restrict__ qh,
                                const _Float16* __restrict__ kh,
                                _Float16* __restrict__ qlh,
                                _Float16* __restrict__ klh) {
  long bm = blockIdx.x;              // BH*LM
  int m = bm & (LM - 1);
  long bh = bm >> 8;
  int d = threadIdx.x;               // 64
  const _Float16* qp = qh + (bh * SEQ + (long)m * 32) * DH + d;
  const _Float16* kp = kh + (bh * SEQ + (long)m * 32) * DH + d;
  float sq = 0.f, sk = 0.f;
#pragma unroll
  for (int l = 0; l < 32; ++l) {
    sq += (float)qp[l * DH];
    sk += (float)kp[l * DH];
  }
  qlh[(bh * LM + m) * DH + d] = (_Float16)(sq * (1.0f / 32));
  klh[(bh * LM + m) * DH + d] = (_Float16)(sk * (1.0f / 32));
}

__global__ void softmax_row256(float* __restrict__ a) {
  __shared__ float red[256];
  long row = blockIdx.x;
  float* p = a + row * 256;
  int t = threadIdx.x;
  float v = p[t];
  red[t] = v;
  __syncthreads();
  for (int s = 128; s > 0; s >>= 1) {
    if (t < s) red[t] = fmaxf(red[t], red[t + s]);
    __syncthreads();
  }
  float mx = red[0];
  __syncthreads();
  float e = __expf(v - mx);
  red[t] = e;
  __syncthreads();
  for (int s = 128; s > 0; s >>= 1) {
    if (t < s) red[t] += red[t + s];
    __syncthreads();
  }
  p[t] = e / red[0];
}

__global__ void colsum_gmax(const float* __restrict__ a2, float* __restrict__ gmax) {
  __shared__ float red[256];
  int bh = blockIdx.x, c = threadIdx.x;
  const float* A = a2 + (long)bh * LM * LM;
  float s = 0.f;
  for (int r = 0; r < LM; ++r) s += A[r * LM + c];
  red[c] = s;
  __syncthreads();
  for (int st = 128; st > 0; st >>= 1) {
    if (c < st) red[c] = fmaxf(red[c], red[c + st]);
    __syncthreads();
  }
  if (c == 0) atomicMax((unsigned int*)gmax, __float_as_uint(red[0]));
}

// z0 = a^T / scale  (row-sums of softmax == 1, so scale == max colsum)
__global__ void z_init(const float* __restrict__ a2, const float* __restrict__ gmax,
                       float* __restrict__ z) {
  long i = (long)blockIdx.x * 256 + threadIdx.x;  // BH*LM*LM
  int c = i & 255;
  int r = (int)((i >> 8) & 255);
  long bh = i >> 16;
  z[i] = a2[(bh << 16) + (long)c * 256 + r] / *gmax;
}

__global__ void eye_sub(float cdiag, const float* __restrict__ src,
                        float* __restrict__ dst) {
  long i = (long)blockIdx.x * 256 + threadIdx.x;
  int c = i & 255;
  int r = (int)((i >> 8) & 255);
  dst[i] = (r == c ? cdiag : 0.f) - src[i];
}

// online-softmax row stats over attn3 scores (row length SEQ)
__global__ void rowstats_kernel(const float* __restrict__ s3,
                                float* __restrict__ rmax, float* __restrict__ rinv) {
  __shared__ float rm[256], rs[256];
  long row = blockIdx.x;  // BH*LM
  const float* p = s3 + row * (long)SEQ;
  int t = threadIdx.x;
  float mx = -1e30f, sm = 0.f;
  for (int j = t; j < SEQ; j += 256) {
    float v = p[j];
    float nm = fmaxf(mx, v);
    sm = sm * __expf(mx - nm) + __expf(v - nm);
    mx = nm;
  }
  rm[t] = mx;
  rs[t] = sm;
  __syncthreads();
  for (int s = 128; s > 0; s >>= 1) {
    if (t < s) {
      float m2 = rm[t + s], s2 = rs[t + s];
      float m = fmaxf(rm[t], m2);
      rs[t] = rs[t] * __expf(rm[t] - m) + s2 * __expf(m2 - m);
      rm[t] = m;
    }
    __syncthreads();
  }
  if (t == 0) {
    rmax[row] = rm[0];
    rinv[row] = 1.f / rs[0];
  }
}

// depthwise 33-tap conv over sequence, accumulated into merged-head layout
__global__ void conv_res_kernel(const float* __restrict__ vf,
                                const float* __restrict__ wres,
                                float* __restrict__ oh) {
  long i = (long)blockIdx.x * 256 + threadIdx.x;  // BH*SEQ*DH
  int d = i & 63;
  int n = (int)((i >> 6) & (SEQ - 1));
  long bh = i >> 19;
  int h = (int)(bh & 7);
  int b = (int)(bh >> 3);
  float s = 0.f;
#pragma unroll
  for (int j = 0; j < 33; ++j) {
    int nn = n + j - 16;
    if (nn >= 0 && nn < SEQ) s += wres[h * 33 + j] * vf[(bh * SEQ + nn) * DH + d];
  }
  oh[((long)b * SEQ + n) * DM + h * DH + d] += s;
}

// ------------------------- generic f16 GEMM (C=A@Bt^T) ----------------------
struct EpiStore {
  float* C;
  long ldc, sC;
  DEVFN void operator()(int b, int r, int c, float v) const {
    C[(long)b * sC + (long)r * ldc + c] = v;
  }
};
struct EpiQKV {  // scatter qkv into per-head layout, scale q
  _Float16 *qh, *kh, *vh;
  float* vf;
  DEVFN void operator()(int, int r, int c, float val) const {
    int sec = c >> 9, rem = c & 511;
    int h = rem >> 6, d = rem & 63;
    int bb = r >> 13, n = r & (SEQ - 1);
    long idx = ((((long)bb * NH + h) * SEQ) + n) * DH + d;
    if (sec == 0) qh[idx] = (_Float16)(val * 0.125f);
    else if (sec == 1) kh[idx] = (_Float16)val;
    else { vh[idx] = (_Float16)val; vf[idx] = val; }
  }
};
struct EpiOut {  // y = acc + b_out + x
  float* y;
  const float *bo, *xin;
  DEVFN void operator()(int, int r, int c, float val) const {
    long i = (long)r * DM + c;
    y[i] = val + bo[c] + xin[i];
  }
};

template <class Epi>
__global__ void __launch_bounds__(128)
gemm_bt_f16(const _Float16* __restrict__ A, long sA,
            const _Float16* __restrict__ Bt, long sB, int lda, int ldb, int Kdim,
            Epi epi) {
  __shared__ __align__(16) _Float16 As[64][40];
  __shared__ __align__(16) _Float16 Bs[64][40];
  const int batch = blockIdx.z;
  const _Float16* Ap = A + (long)batch * sA;
  const _Float16* Bp = Bt + (long)batch * sB;
  const int m0 = blockIdx.y * 64, n0 = blockIdx.x * 64;
  const int wave = threadIdx.x >> 5;
  const int wm = (wave & 1) * 32, wn = (wave >> 1) * 32;
  v8f acc[2][2] = {};
  for (int k0 = 0; k0 < Kdim; k0 += 32) {
    for (int i = threadIdx.x; i < 256; i += 128) {
      int r = i >> 2, c8 = (i & 3) * 8;
      *(uint4*)&As[r][c8] = *(const uint4*)&Ap[(long)(m0 + r) * lda + k0 + c8];
      *(uint4*)&Bs[r][c8] = *(const uint4*)&Bp[(long)(n0 + r) * ldb + k0 + c8];
    }
    __syncthreads();
    v16h a0 = load_a(&As[wm][0], 40);
    v16h a1 = load_a(&As[wm + 16][0], 40);
    v16h b0 = load_bt(&Bs[wn][0], 40);
    v16h b1 = load_bt(&Bs[wn + 16][0], 40);
    acc[0][0] = wmma16(a0, b0, acc[0][0]);
    acc[0][1] = wmma16(a0, b1, acc[0][1]);
    acc[1][0] = wmma16(a1, b0, acc[1][0]);
    acc[1][1] = wmma16(a1, b1, acc[1][1]);
    __syncthreads();
  }
  const int lane = threadIdx.x & 31;
  const int cn = lane & 15, cm = (lane >> 4) * 8;
#pragma unroll
  for (int i = 0; i < 2; ++i)
#pragma unroll
    for (int j = 0; j < 2; ++j)
#pragma unroll
      for (int e = 0; e < 8; ++e)
        epi(batch, m0 + wm + i * 16 + cm + e, n0 + wn + j * 16 + cn, acc[i][j][e]);
}

// --------------------- fp32 GEMM (C = alpha * A@B), batched -----------------
__global__ void __launch_bounds__(128)
gemm_nn_f32(const float* __restrict__ A, long sA, const float* __restrict__ B,
            long sB, float* __restrict__ C, long sC, int lda, int ldb, int ldc,
            int Kdim, float alpha) {
  __shared__ __align__(16) float As[64][36];
  __shared__ __align__(16) float Bs[32][68];
  const int batch = blockIdx.z;
  const float* Ap = A + (long)batch * sA;
  const float* Bp = B + (long)batch * sB;
  const int m0 = blockIdx.y * 64, n0 = blockIdx.x * 64;
  const int wave = threadIdx.x >> 5;
  const int wm = (wave & 1) * 32, wn = (wave >> 1) * 32;
  const int lane = threadIdx.x & 31;
  const int ml = lane & 15, nl = lane & 15, hi2 = (lane >> 4) * 2;
  v8f acc[2][2] = {};
  for (int k0 = 0; k0 < Kdim; k0 += 32) {
    for (int i = threadIdx.x; i < 512; i += 128) {
      { int r = i >> 3, c4 = (i & 7) * 4;
        *(float4*)&As[r][c4] = *(const float4*)&Ap[(long)(m0 + r) * lda + k0 + c4]; }
      { int r = i >> 4, c4 = (i & 15) * 4;
        *(float4*)&Bs[r][c4] = *(const float4*)&Bp[(long)(k0 + r) * ldb + n0 + c4]; }
    }
    __syncthreads();
#pragma unroll
    for (int kk = 0; kk < 32; kk += 4) {
      v2f a0, a1, b0, b1;
      a0[0] = As[wm + ml][kk + hi2];       a0[1] = As[wm + ml][kk + hi2 + 1];
      a1[0] = As[wm + 16 + ml][kk + hi2];  a1[1] = As[wm + 16 + ml][kk + hi2 + 1];
      b0[0] = Bs[kk + hi2][wn + nl];       b0[1] = Bs[kk + hi2 + 1][wn + nl];
      b1[0] = Bs[kk + hi2][wn + 16 + nl];  b1[1] = Bs[kk + hi2 + 1][wn + 16 + nl];
      acc[0][0] = wmma4(a0, b0, acc[0][0]);
      acc[0][1] = wmma4(a0, b1, acc[0][1]);
      acc[1][0] = wmma4(a1, b0, acc[1][0]);
      acc[1][1] = wmma4(a1, b1, acc[1][1]);
    }
    __syncthreads();
  }
  const int cn = lane & 15, cm = (lane >> 4) * 8;
#pragma unroll
  for (int i = 0; i < 2; ++i)
#pragma unroll
    for (int j = 0; j < 2; ++j)
#pragma unroll
      for (int e = 0; e < 8; ++e)
        C[(long)batch * sC + (long)(m0 + wm + i * 16 + cm + e) * ldc +
          (n0 + wn + j * 16 + cn)] = alpha * acc[i][j][e];
}

// -------- U = softmax(attn3-scores) @ V, P generated on the fly (f16) -------
__global__ void __launch_bounds__(128)
gemm_pv(const float* __restrict__ s3, const float* __restrict__ rmax,
        const float* __restrict__ rinv, const _Float16* __restrict__ vh,
        float* __restrict__ uf) {
  __shared__ __align__(16) _Float16 As[64][40];
  __shared__ __align__(16) _Float16 Bs[32][72];
  const int batch = blockIdx.z;
  const float* S = s3 + (long)batch * LM * SEQ;
  const _Float16* V = vh + (long)batch * SEQ * DH;
  const int m0 = blockIdx.y * 64;
  const int wave = threadIdx.x >> 5;
  const int wm = (wave & 1) * 32, wn = (wave >> 1) * 32;
  v8f acc[2][2] = {};
  for (int k0 = 0; k0 < SEQ; k0 += 32) {
    for (int i = threadIdx.x; i < 2048; i += 128) {
      int r = i >> 5, c = i & 31;
      int row = m0 + r;
      float s = S[(long)row * SEQ + k0 + c];
      As[r][c] = (_Float16)(__expf(s - rmax[batch * LM + row]) *
                            rinv[batch * LM + row]);
    }
    for (int i = threadIdx.x; i < 256; i += 128) {
      int r = i >> 3, c8 = (i & 7) * 8;
      *(uint4*)&Bs[r][c8] = *(const uint4*)&V[(long)(k0 + r) * DH + c8];
    }
    __syncthreads();
    v16h a0 = load_a(&As[wm][0], 40);
    v16h a1 = load_a(&As[wm + 16][0], 40);
    v16h b0 = load_b(&Bs[0][wn], 72);
    v16h b1 = load_b(&Bs[0][wn + 16], 72);
    acc[0][0] = wmma16(a0, b0, acc[0][0]);
    acc[0][1] = wmma16(a0, b1, acc[0][1]);
    acc[1][0] = wmma16(a1, b0, acc[1][0]);
    acc[1][1] = wmma16(a1, b1, acc[1][1]);
    __syncthreads();
  }
  const int lane = threadIdx.x & 31;
  const int cn = lane & 15, cm = (lane >> 4) * 8;
#pragma unroll
  for (int i = 0; i < 2; ++i)
#pragma unroll
    for (int j = 0; j < 2; ++j)
#pragma unroll
      for (int e = 0; e < 8; ++e)
        uf[(long)batch * LM * DH + (long)(m0 + wm + i * 16 + cm + e) * DH +
           (wn + j * 16 + cn)] = acc[i][j][e];
}

// --- fused attn1: per 32 q-rows: s=q@kl^T (WMMA) -> exact softmax -> P@W ----
__global__ void __launch_bounds__(256)
attn1_fused(const _Float16* __restrict__ qh, const _Float16* __restrict__ klh,
            const _Float16* __restrict__ wh, float* __restrict__ oh) {
  __shared__ __align__(16) _Float16 Qs[32][64];
  __shared__ __align__(16) _Float16 Ks[LM][64];
  __shared__ _Float16 Ps[32][264];
  const int bh = blockIdx.y, b = bh >> 3, h = bh & 7;
  const int n0 = blockIdx.x * 32;
  const _Float16* q  = qh + ((long)bh * SEQ + n0) * DH;
  const _Float16* kl = klh + (long)bh * LM * DH;
  const _Float16* w  = wh + (long)bh * LM * DH;
  const int t = threadIdx.x, wave = t >> 5, lane = t & 31;
  {
    int r = t >> 3, c8 = (t & 7) * 8;
    *(uint4*)&Qs[r][c8] = *(const uint4*)&q[r * 64 + c8];
  }
  for (int i = t; i < 2048; i += 256) {
    int r = i >> 3, c8 = (i & 7) * 8;
    *(uint4*)&Ks[r][c8] = *(const uint4*)&kl[r * 64 + c8];
  }
  __syncthreads();
  {  // scores: wave owns a 32-column strip
    const int wn = wave * 32;
    v8f acc[2][2] = {};
#pragma unroll
    for (int k0 = 0; k0 < 64; k0 += 32) {
      v16h a0 = load_a(&Qs[0][k0], 64);
      v16h a1 = load_a(&Qs[16][k0], 64);
      v16h b0 = load_bt(&Ks[wn][k0], 64);
      v16h b1 = load_bt(&Ks[wn + 16][k0], 64);
      acc[0][0] = wmma16(a0, b0, acc[0][0]);
      acc[0][1] = wmma16(a0, b1, acc[0][1]);
      acc[1][0] = wmma16(a1, b0, acc[1][0]);
      acc[1][1] = wmma16(a1, b1, acc[1][1]);
    }
    const int cn = lane & 15, cm = (lane >> 4) * 8;
#pragma unroll
    for (int i = 0; i < 2; ++i)
#pragma unroll
      for (int j = 0; j < 2; ++j)
#pragma unroll
        for (int e = 0; e < 8; ++e)
          Ps[i * 16 + cm + e][wn + j * 16 + cn] = (_Float16)acc[i][j][e];
  }
  __syncthreads();
  // exact softmax (row length 256): one wave per row, 4 rows each
  for (int rr = 0; rr < 4; ++rr) {
    int r = wave + rr * 8;
    float vals[8];
    float mx = -1e30f;
#pragma unroll
    for (int j = 0; j < 8; ++j) {
      vals[j] = (float)Ps[r][lane + j * 32];
      mx = fmaxf(mx, vals[j]);
    }
#pragma unroll
    for (int m = 16; m > 0; m >>= 1) mx = fmaxf(mx, __shfl_xor(mx, m, 32));
    float s = 0.f;
#pragma unroll
    for (int j = 0; j < 8; ++j) {
      vals[j] = __expf(vals[j] - mx);
      s += vals[j];
    }
#pragma unroll
    for (int m = 16; m > 0; m >>= 1) s += __shfl_xor(s, m, 32);
    float inv = 1.f / s;
#pragma unroll
    for (int j = 0; j < 8; ++j)
      Ps[r][lane + j * 32] = (_Float16)(vals[j] * inv);
  }
  __syncthreads();
  // out = P @ W   ([32,256]@[256,64]); 8 waves = 2x4 16x16 tiles
  const int mt = wave & 1, nt = wave >> 1;
  v8f o = {};
  for (int k0 = 0; k0 < LM; k0 += 32) {
    v16h a = load_a(&Ps[mt * 16][k0], 264);
    v16h bf = load_b(&w[(long)k0 * DH + nt * 16], DH);
    o = wmma16(a, bf, o);
  }
  const int cn = lane & 15, cm = (lane >> 4) * 8;
#pragma unroll
  for (int e = 0; e < 8; ++e) {
    int n = n0 + mt * 16 + cm + e;
    oh[((long)b * SEQ + n) * DM + h * DH + nt * 16 + cn] = o[e];
  }
}

// ------------------------------- host launch --------------------------------
extern "C" void kernel_launch(void* const* d_in, const int* in_sizes, int n_in,
                              void* d_out, int out_size, void* d_ws,
                              size_t ws_size, hipStream_t stream) {
  (void)in_sizes; (void)n_in; (void)out_size; (void)ws_size;
  const float* x     = (const float*)d_in[0];
  const float* gamma = (const float*)d_in[1];
  const float* beta  = (const float*)d_in[2];
  const float* wqkv  = (const float*)d_in[3];
  const float* wout  = (const float*)d_in[4];
  const float* bout  = (const float*)d_in[5];
  const float* wres  = (const float*)d_in[6];
  float* y = (float*)d_out;

  char* ws = (char*)d_ws;
  size_t off = 0;
  auto alloc = [&](size_t bytes) {
    void* p = ws + off;
    off = (off + bytes + 255) & ~(size_t)255;
    return p;
  };
  _Float16* XNH   = (_Float16*)alloc((size_t)BN * DM * 2);
  _Float16* WQKVH = (_Float16*)alloc((size_t)3 * DM * DM * 2);
  _Float16* WOUTH = (_Float16*)alloc((size_t)DM * DM * 2);
  _Float16* QH    = (_Float16*)alloc((size_t)BH * SEQ * DH * 2);
  _Float16* KH    = (_Float16*)alloc((size_t)BH * SEQ * DH * 2);
  _Float16* VH    = (_Float16*)alloc((size_t)BH * SEQ * DH * 2);
  float*    VF    = (float*)alloc((size_t)BH * SEQ * DH * 4);
  _Float16* QLH   = (_Float16*)alloc((size_t)BH * LM * DH * 2);
  _Float16* KLH   = (_Float16*)alloc((size_t)BH * LM * DH * 2);
  float*    A2    = (float*)alloc((size_t)BH * LM * LM * 4);
  float*    ZB    = (float*)alloc((size_t)BH * LM * LM * 4);
  float*    AZ    = (float*)alloc((size_t)BH * LM * LM * 4);
  float*    TB    = (float*)alloc((size_t)BH * LM * LM * 4);
  float*    ZT    = (float*)alloc((size_t)BH * LM * LM * 4);
  float*    S3    = (float*)alloc((size_t)BH * LM * SEQ * 4);
  float*    RMAX  = (float*)alloc((size_t)BH * LM * 4);
  float*    RINV  = (float*)alloc((size_t)BH * LM * 4);
  float*    UF    = (float*)alloc((size_t)BH * LM * DH * 4);
  float*    WF    = (float*)alloc((size_t)BH * LM * DH * 4);
  _Float16* WHh   = (_Float16*)alloc((size_t)BH * LM * DH * 2);
  float*    OH    = (float*)alloc((size_t)BN * DM * 4);
  _Float16* OHH   = (_Float16*)alloc((size_t)BN * DM * 2);
  float*    GMAX  = (float*)alloc(256);

  const long MM2 = (long)LM * LM;  // 65536

  // weights -> f16
  f2h_kernel<<<(3 * DM * DM + 255) / 256, 256, 0, stream>>>(wqkv, WQKVH,
                                                            (long)3 * DM * DM);
  f2h_kernel<<<(DM * DM + 255) / 256, 256, 0, stream>>>(wout, WOUTH,
                                                        (long)DM * DM);
  // LayerNorm
  layernorm_kernel<<<BN, 256, 0, stream>>>(x, gamma, beta, XNH);
  // QKV projection (scatter to per-head q/k/v, q scaled)
  gemm_bt_f16<EpiQKV><<<dim3(24, 256, 1), 128, 0, stream>>>(
      XNH, 0, WQKVH, 0, DM, DM, DM, EpiQKV{QH, KH, VH, VF});
  // landmarks (mean pool by 32)
  landmark_kernel<<<BH * LM, 64, 0, stream>>>(QH, KH, QLH, KLH);
  // attn2 scores + softmax
  gemm_bt_f16<EpiStore><<<dim3(4, 4, BH), 128, 0, stream>>>(
      QLH, (long)LM * DH, KLH, (long)LM * DH, DH, DH, DH, EpiStore{A2, LM, MM2});
  softmax_row256<<<BH * LM, 256, 0, stream>>>(A2);
  // pinv scale: global max colsum (row sums are exactly 1)
  hipMemsetAsync(GMAX, 0, 4, stream);
  colsum_gmax<<<BH, 256, 0, stream>>>(A2, GMAX);
  z_init<<<(int)(BH * MM2 / 256), 256, 0, stream>>>(A2, GMAX, ZB);
  // Newton-Schulz pseudoinverse, 6 iterations, fp32 WMMA
  float* zc = ZB;
  float* zt = ZT;
  for (int it = 0; it < 6; ++it) {
    gemm_nn_f32<<<dim3(4, 4, BH), 128, 0, stream>>>(A2, MM2, zc, MM2, AZ, MM2,
                                                    LM, LM, LM, LM, 1.0f);
    eye_sub<<<(int)(BH * MM2 / 256), 256, 0, stream>>>(7.f, AZ, TB);
    gemm_nn_f32<<<dim3(4, 4, BH), 128, 0, stream>>>(AZ, MM2, TB, MM2, zt, MM2,
                                                    LM, LM, LM, LM, 1.0f);
    eye_sub<<<(int)(BH * MM2 / 256), 256, 0, stream>>>(15.f, zt, TB);
    gemm_nn_f32<<<dim3(4, 4, BH), 128, 0, stream>>>(AZ, MM2, TB, MM2, zt, MM2,
                                                    LM, LM, LM, LM, 1.0f);
    eye_sub<<<(int)(BH * MM2 / 256), 256, 0, stream>>>(13.f, zt, TB);
    gemm_nn_f32<<<dim3(4, 4, BH), 128, 0, stream>>>(zc, MM2, TB, MM2, zt, MM2,
                                                    LM, LM, LM, LM, 0.25f);
    float* tmp = zc; zc = zt; zt = tmp;
  }
  // attn3 scores: ql @ k^T  -> [BH, LM, SEQ]
  gemm_bt_f16<EpiStore><<<dim3(SEQ / 64, LM / 64, BH), 128, 0, stream>>>(
      QLH, (long)LM * DH, KH, (long)SEQ * DH, DH, DH, DH,
      EpiStore{S3, SEQ, (long)LM * SEQ});
  rowstats_kernel<<<BH * LM, 256, 0, stream>>>(S3, RMAX, RINV);
  // U = softmax(s3) @ V
  gemm_pv<<<dim3(1, LM / 64, BH), 128, 0, stream>>>(S3, RMAX, RINV, VH, UF);
  // W = Z @ U  (fp32)
  gemm_nn_f32<<<dim3(1, 4, BH), 128, 0, stream>>>(zc, MM2, UF, (long)LM * DH, WF,
                                                  (long)LM * DH, LM, DH, DH, LM,
                                                  1.0f);
  f2h_kernel<<<(BH * LM * DH + 255) / 256, 256, 0, stream>>>(WF, WHh,
                                                             (long)BH * LM * DH);
  // fused attn1: softmax(q @ kl^T) @ W -> merged-head output
  attn1_fused<<<dim3(SEQ / 32, BH), 256, 0, stream>>>(QH, KLH, WHh, OH);
  // depthwise conv residual
  conv_res_kernel<<<(int)((long)BH * SEQ * DH / 256), 256, 0, stream>>>(VF, wres,
                                                                        OH);
  // output projection + bias + residual
  f2h_kernel<<<(BN * DM + 255) / 256, 256, 0, stream>>>(OH, OHH, (long)BN * DM);
  gemm_bt_f16<EpiOut><<<dim3(8, 256, 1), 128, 0, stream>>>(
      OHH, 0, WOUTH, 0, DM, DM, DM, EpiOut{y, bout, x});
}
